// Encoder_23957327577491
// MI455X (gfx1250) — compile-verified, matrix operand order checked by reading, not run on hardware
//
#include <hip/hip_runtime.h>
#include <hip/hip_bf16.h>
#include <cstdint>

// ---------------- problem constants (match reference) ----------------
constexpr int CB   = 2;          // batch
constexpr int CS   = 2048;       // sequence length
constexpr int CF   = 64;         // input features
constexpr int CE   = 512;        // embed dim
constexpr int CH   = 8;          // heads
constexpr int CL   = 4;          // layers
constexpr int CEXP = 4;          // FF expansion
constexpr int CDH  = CE / CH;    // 64
constexpr int CT   = CB * CS;    // 4096 tokens
constexpr float CEPS = 1e-5f;

typedef __attribute__((ext_vector_type(16))) __bf16 v16bf;
typedef __attribute__((ext_vector_type(8)))  float  v8f;

__device__ __forceinline__ v8f wmma_bf16(v16bf a, v16bf b, v8f c) {
  // D = A(16x32) * B(32x16) + C, fp32 accumulate
  return __builtin_amdgcn_wmma_f32_16x16x32_bf16(false, a, false, b,
                                                 (short)0, c, false, false);
}

// Async 16-byte copy global->LDS (per lane), tracked by ASYNCcnt.
__device__ __forceinline__ void async_ld16(void* lds, const void* g) {
  asm volatile("global_load_async_to_lds_b128 %0, %1, off"
               :
               : "v"((unsigned)(size_t)lds), "v"(g)
               : "memory");
}
__device__ __forceinline__ void wait_async_le4() {
  asm volatile("s_wait_asynccnt 0x4" ::: "memory");
}
__device__ __forceinline__ void wait_async_le8() {
  asm volatile("s_wait_asynccnt 0x8" ::: "memory");
}
__device__ __forceinline__ void wait_async_0() {
  asm volatile("s_wait_asynccnt 0x0" ::: "memory");
}

// Unified fragment loader (16x32 bf16 A-layout; also serves B-fragments when
// the source tile is stored transposed, i.e. fragment rows are contiguous):
// lanes 0-15: row m, K in {0..7,16..23}; lanes 16-31: row m, K in {8..15,24..31};
// each VGPR packs (k,k+1) -> two contiguous 16-byte LDS reads (ds_load_b128).
__device__ __forceinline__ v16bf load_frag(const __bf16* base, int hf, int m, int ld) {
  const __bf16* row = base + m * ld;
  v16bf f;
#pragma unroll
  for (int v = 0; v < 8; ++v) {
    const int kb = (v < 4) ? (2 * v) : (16 + 2 * (v - 4));
    const int k  = kb + 8 * hf;
    f[2 * v]     = row[k];
    f[2 * v + 1] = row[k + 1];
  }
  return f;
}

// ---------------- fp32 -> bf16 conversion (row-major copy) ----------------
__global__ __launch_bounds__(256) void k_cvt_bf16(const float* __restrict__ in,
                                                  __bf16* __restrict__ out, int n) {
  int i = (blockIdx.x * blockDim.x + threadIdx.x) * 4;
  if (i + 3 < n) {
    float4 v = *(const float4*)(in + i);
    out[i]     = (__bf16)v.x;
    out[i + 1] = (__bf16)v.y;
    out[i + 2] = (__bf16)v.z;
    out[i + 3] = (__bf16)v.w;
  } else {
    for (; i < n; ++i) out[i] = (__bf16)in[i];
  }
}

// ---------------- fp32 -> bf16 convert + transpose: out[N][K] = in[K][N]^T ----
__global__ __launch_bounds__(256) void k_cvt_transpose(
    const float* __restrict__ in, __bf16* __restrict__ out, int K, int N) {
  __shared__ float tile[64][65];
  const int k0 = blockIdx.y * 64;
  const int n0 = blockIdx.x * 64;
  const size_t slice = (size_t)blockIdx.z * K * N;
  const int tx = threadIdx.x & 63;
  const int ty = threadIdx.x >> 6;          // 0..3
#pragma unroll
  for (int i = 0; i < 16; ++i)
    tile[ty + i * 4][tx] = in[slice + (size_t)(k0 + ty + i * 4) * N + n0 + tx];
  __syncthreads();
#pragma unroll
  for (int i = 0; i < 16; ++i)
    out[slice + (size_t)(n0 + ty + i * 4) * K + k0 + tx] =
        (__bf16)tile[tx][ty + i * 4];
}

// ---------------- tiled WMMA GEMM: C[M,N] = A[M,K] @ Bt[N,K]^T + bias --------
// Double-buffered LDS fed by async global->LDS copies; WMMA overlaps the loads.
constexpr int GBM = 128, GBN = 128, GBK = 32;
constexpr int LDA = GBK + 8;    // 40 bf16 elems (padded)
constexpr int LDBT = GBK + 8;   // 40 (B tile stored [n][k])

template <bool RELU, bool WF, bool WB>
__global__ __launch_bounds__(256) void k_gemm_bf16(
    const __bf16* __restrict__ A, const __bf16* __restrict__ Bt,
    const float* __restrict__ bias,
    float* __restrict__ outF, __bf16* __restrict__ outB,
    int M, int N, int K)
{
  __shared__ __align__(16) __bf16 As[2][GBM * LDA];
  __shared__ __align__(16) __bf16 Bs[2][GBN * LDBT];

  const int tid  = threadIdx.x;
  const int lane = tid & 31;
  const int wave = tid >> 5;          // 8 waves
  const int wr   = wave >> 1;         // 0..3: 32-row group
  const int wc   = wave & 1;          // 0..1: 64-col group
  const int hf   = lane >> 4, ln = lane & 15;
  const int bm   = blockIdx.y * GBM;
  const int bn   = blockIdx.x * GBN;

  // issue one tile's async copies (4 instructions/wave -> ASYNCcnt += 4)
  auto stage = [&](int buf, int k0) {
#pragma unroll
    for (int i = 0; i < 2; ++i) {
      const int c = tid + i * 256;
      const int r = c >> 2, col = (c & 3) * 8;
      async_ld16(&As[buf][r * LDA + col], A + (size_t)(bm + r) * K + k0 + col);
      async_ld16(&Bs[buf][r * LDBT + col], Bt + (size_t)(bn + r) * K + k0 + col);
    }
  };

  v8f acc[2][4];
#pragma unroll
  for (int i = 0; i < 2; ++i)
#pragma unroll
    for (int j = 0; j < 4; ++j)
#pragma unroll
      for (int r = 0; r < 8; ++r) acc[i][j][r] = 0.f;

  const int ntile = K / GBK;
  stage(0, 0);
  for (int it = 0; it < ntile; ++it) {
    if (it + 1 < ntile) {
      stage((it + 1) & 1, (it + 1) * GBK);
      wait_async_le4();     // newest 4 = tile it+1; in-order => tile it resident
    } else {
      wait_async_0();
    }
    __syncthreads();        // all waves' copies for tile it complete

    const __bf16* Ab = As[it & 1];
    const __bf16* Bb = Bs[it & 1];
    v16bf af[2], bfr[4];
#pragma unroll
    for (int i = 0; i < 2; ++i)
      af[i] = load_frag(&Ab[(wr * 32 + i * 16) * LDA], hf, ln, LDA);
#pragma unroll
    for (int j = 0; j < 4; ++j)
      bfr[j] = load_frag(&Bb[(wc * 64 + j * 16) * LDBT], hf, ln, LDBT);
#pragma unroll
    for (int i = 0; i < 2; ++i)
#pragma unroll
      for (int j = 0; j < 4; ++j)
        acc[i][j] = wmma_bf16(af[i], bfr[j], acc[i][j]);

    __syncthreads();        // readers done before tile it+2 overwrites buffer
  }

  // epilogue: C layout = lanes 0-15 rows 0..7 (VGPR r), lanes 16-31 rows 8..15
#pragma unroll
  for (int i = 0; i < 2; ++i)
#pragma unroll
    for (int j = 0; j < 4; ++j) {
      const int col = bn + wc * 64 + j * 16 + ln;
      const float bv = bias[col];
#pragma unroll
      for (int r = 0; r < 8; ++r) {
        const int row = bm + wr * 32 + i * 16 + hf * 8 + r;
        float v = acc[i][j][r] + bv;
        if constexpr (RELU) v = fmaxf(v, 0.f);
        if constexpr (WF) outF[(size_t)row * N + col] = v;
        if constexpr (WB) outB[(size_t)row * N + col] = (__bf16)v;
      }
    }
}

// ---------------- embedding positional add ----------------
__global__ __launch_bounds__(256) void k_addpos(const float* __restrict__ emb,
                                                const float* __restrict__ pos,
                                                float* __restrict__ xf,
                                                __bf16* __restrict__ xb) {
  const int t = blockIdx.x;
  const int s = t & (CS - 1);
#pragma unroll
  for (int i = 0; i < 2; ++i) {
    const int e = threadIdx.x + i * 256;
    const float v = emb[(size_t)t * CE + e] + pos[(size_t)s * CE + e];
    xf[(size_t)t * CE + e] = v;
    xb[(size_t)t * CE + e] = (__bf16)v;
  }
}

// ---------------- residual + LayerNorm ----------------
__global__ __launch_bounds__(256) void k_ln(const float* __restrict__ xres,
                                            const float* __restrict__ delta,
                                            const float* __restrict__ g,
                                            const float* __restrict__ bta,
                                            float* __restrict__ outF,
                                            __bf16* __restrict__ outB) {
  __shared__ float rs[8], rs2[8];
  const int t = blockIdx.x, tid = threadIdx.x;
  const int lane = tid & 31, wave = tid >> 5;
  const size_t base = (size_t)t * CE;
  const float v0 = xres[base + tid] + delta[base + tid];
  const float v1 = xres[base + 256 + tid] + delta[base + 256 + tid];
  float s = v0 + v1, s2 = v0 * v0 + v1 * v1;
#pragma unroll
  for (int m = 16; m >= 1; m >>= 1) {
    s  += __shfl_xor(s, m, 32);
    s2 += __shfl_xor(s2, m, 32);
  }
  if (lane == 0) { rs[wave] = s; rs2[wave] = s2; }
  __syncthreads();
  float tot = 0.f, tot2 = 0.f;
#pragma unroll
  for (int i = 0; i < 8; ++i) { tot += rs[i]; tot2 += rs2[i]; }
  const float mean = tot * (1.f / CE);
  const float var  = tot2 * (1.f / CE) - mean * mean;
  const float rstd = rsqrtf(var + CEPS);
  const float y0 = (v0 - mean) * rstd * g[tid] + bta[tid];
  const float y1 = (v1 - mean) * rstd * g[tid + 256] + bta[tid + 256];
  outF[base + tid] = y0;           outF[base + 256 + tid] = y1;
  outB[base + tid] = (__bf16)y0;   outB[base + 256 + tid] = (__bf16)y1;
}

// ---------------- per-(b,h) V transpose: vt[(b*H+h)][d][s] = v[b*S+s][h*64+d] --
__global__ __launch_bounds__(256) void k_vtrans(
    const __bf16* __restrict__ vin, __bf16* __restrict__ vt) {
  __shared__ __align__(16) __bf16 tile[64][72];
  const int s0 = blockIdx.x * 64;
  const int bh = blockIdx.y;                 // b*CH + h
  const int bb = bh >> 3, h = bh & 7;        // CH == 8
  const int tid = threadIdx.x;
#pragma unroll
  for (int i = 0; i < 2; ++i) {
    const int c = tid + i * 256;
    const int r = c >> 3, col = (c & 7) * 8;  // r: s-row, col: d
    *(uint4*)&tile[r][col] =
        *(const uint4*)(vin + ((size_t)bb * CS + s0 + r) * CE + h * CDH + col);
  }
  __syncthreads();
#pragma unroll
  for (int i = 0; i < 2; ++i) {
    const int c = tid + i * 256;
    const int d = c >> 3, scol = (c & 7) * 8;
    union { uint4 u; __bf16 e[8]; } pk;
#pragma unroll
    for (int e = 0; e < 8; ++e) pk.e[e] = tile[scol + e][d];
    *(uint4*)(vt + ((size_t)bh * CDH + d) * CS + s0 + scol) = pk.u;
  }
}

// ---------------- flash attention (bf16 WMMA, exp2-domain fp32 softmax) ------
// 64-key blocks halve the per-key softmax fixed cost; K/V tiles double-buffered
// via async global->LDS copies so staging hides behind the 16 WMMAs per block.
// log2(e) is folded into the Q pre-scale so every exponential is a bare
// v_exp_f32 (softmax is invariant under the base change).
constexpr int AQ  = 64;          // query rows per block (4 waves x 16)
constexpr int AK  = 64;          // keys per iteration
constexpr int KLD = CDH + 8;     // 72  (K tile [key][d])
constexpr int VLD = AK + 8;      // 72  (V^T tile [d][key])
constexpr int PLD = AK + 8;      // 72  (P tile [row][key])

__global__ __launch_bounds__(128) void k_attn(
    const __bf16* __restrict__ Q, const __bf16* __restrict__ Kb,
    const __bf16* __restrict__ VT, const int* __restrict__ mask,
    __bf16* __restrict__ O)
{
  __shared__ __align__(16) __bf16 Ks[2][AK * KLD];
  __shared__ __align__(16) __bf16 VTs[2][CDH * VLD];
  __shared__ __align__(16) __bf16 Ps[4][16 * PLD];  // per-wave P reshaping tile

  const int qb = blockIdx.x, h = blockIdx.y, b = blockIdx.z;
  const int tid = threadIdx.x, lane = tid & 31, wave = tid >> 5;
  const int hf = lane >> 4, ln = lane & 15;
  const size_t tok0 = (size_t)b * CS;
  const int row0 = qb * AQ + wave * 16;
  const int coff = h * CDH;
  const int bh = b * CH + h;
  // (1/sqrt(E)) * log2(e): scores land directly in the exp2 domain
  const float scale = 0.04419417382415922f * 1.4426950408889634f;

  // issue one 64-key block's async copies (8 instructions/wave)
  auto stage = [&](int buf, int kb0) {
#pragma unroll
    for (int i = 0; i < 4; ++i) {
      const int c = tid + i * 128;
      const int r = c >> 3, col = (c & 7) * 8;
      async_ld16(&Ks[buf][r * KLD + col],
                 Kb + (tok0 + kb0 + r) * (size_t)CE + coff + col);
      async_ld16(&VTs[buf][r * VLD + col],
                 VT + ((size_t)bh * CDH + r) * CS + kb0 + col);
    }
  };

  // Q A-fragments, pre-scaled (folded out of the inner loop)
  v16bf qf[2];
  {
    const __bf16* qrow = Q + (tok0 + row0 + ln) * (size_t)CE + coff;
#pragma unroll
    for (int t = 0; t < 2; ++t)
#pragma unroll
      for (int v = 0; v < 8; ++v) {
        const int kb = (v < 4) ? (2 * v) : (16 + 2 * (v - 4));
        const int d  = t * 32 + kb + 8 * hf;
        qf[t][2 * v]     = (__bf16)((float)qrow[d] * scale);
        qf[t][2 * v + 1] = (__bf16)((float)qrow[d + 1] * scale);
      }
  }

  float mrow[8], lrow[8];
  v8f oacc[4];
#pragma unroll
  for (int r = 0; r < 8; ++r) { mrow[r] = -3.0e38f; lrow[r] = 0.f; }
#pragma unroll
  for (int j = 0; j < 4; ++j)
#pragma unroll
    for (int r = 0; r < 8; ++r) oacc[j][r] = 0.f;

  constexpr int NBLK = CS / AK;     // 32
  stage(0, 0);
  for (int it = 0; it < NBLK; ++it) {
    if (it + 1 < NBLK) {
      stage((it + 1) & 1, (it + 1) * AK);
      wait_async_le8();   // newest 8 = block it+1; in-order => block it resident
    } else {
      wait_async_0();
    }
    __syncthreads();

    const __bf16* Kt = Ks[it & 1];
    const __bf16* Vt = VTs[it & 1];
    const int kb0 = it * AK;

    // scores S[16 x 64] = (Q*scale) @ K^T  (4 key tiles x 2 Dh steps)
    v8f sc[4];
#pragma unroll
    for (int nt = 0; nt < 4; ++nt) {
#pragma unroll
      for (int r = 0; r < 8; ++r) sc[nt][r] = 0.f;
#pragma unroll
      for (int t = 0; t < 2; ++t) {
        v16bf kf = load_frag(&Kt[(nt * 16) * KLD + t * 32], hf, ln, KLD);
        sc[nt] = wmma_bf16(qf[t], kf, sc[nt]);
      }
    }
    // mask (reference: mask==0 -> -1e20, applied before softmax)
#pragma unroll
    for (int nt = 0; nt < 4; ++nt) {
      const int key = kb0 + nt * 16 + ln;
      const bool dead = (mask[(size_t)b * CS + key] == 0);
#pragma unroll
      for (int r = 0; r < 8; ++r)
        sc[nt][r] = dead ? -1.0e20f : sc[nt][r];
    }
    // online softmax (exp2 domain): row r lives across 16 lanes of each half
#pragma unroll
    for (int r = 0; r < 8; ++r) {
      float smax = fmaxf(fmaxf(sc[0][r], sc[1][r]), fmaxf(sc[2][r], sc[3][r]));
#pragma unroll
      for (int m = 8; m >= 1; m >>= 1) smax = fmaxf(smax, __shfl_xor(smax, m, 32));
      const float mnew  = fmaxf(mrow[r], smax);
      const float alpha = exp2f(mrow[r] - mnew);
      float p[4];
#pragma unroll
      for (int nt = 0; nt < 4; ++nt) p[nt] = exp2f(sc[nt][r] - mnew);
      float psum = (p[0] + p[1]) + (p[2] + p[3]);
#pragma unroll
      for (int m = 8; m >= 1; m >>= 1) psum += __shfl_xor(psum, m, 32);
      lrow[r] = lrow[r] * alpha + psum;
      mrow[r] = mnew;
#pragma unroll
      for (int nt = 0; nt < 4; ++nt) sc[nt][r] = p[nt];
#pragma unroll
      for (int j = 0; j < 4; ++j) oacc[j][r] *= alpha;
    }
    // P: C-layout -> A-layout via per-wave LDS tile [16 rows x 64 keys]
    __bf16* pw = &Ps[wave][0];
#pragma unroll
    for (int nt = 0; nt < 4; ++nt)
#pragma unroll
      for (int r = 0; r < 8; ++r)
        pw[(r + 8 * hf) * PLD + nt * 16 + ln] = (__bf16)sc[nt][r];
    v16bf pf[2];
#pragma unroll
    for (int t = 0; t < 2; ++t)
      pf[t] = load_frag(&pw[t * 32], hf, ln, PLD);
    // O += P[16x64] @ V[64x64]  (2 key steps x 4 d tiles)
#pragma unroll
    for (int t = 0; t < 2; ++t)
#pragma unroll
      for (int j = 0; j < 4; ++j) {
        v16bf vf = load_frag(&Vt[(j * 16) * VLD + t * 32], hf, ln, VLD);
        oacc[j] = wmma_bf16(pf[t], vf, oacc[j]);
      }

    __syncthreads();   // readers done before block it+2 overwrites buffer
  }

  // normalize (one reciprocal per row) and store bf16 for the Wo GEMM
#pragma unroll
  for (int r = 0; r < 8; ++r) {
    const float rl = 1.0f / lrow[r];
    const int row = row0 + r + 8 * hf;
#pragma unroll
    for (int j = 0; j < 4; ++j) {
      const float val = oacc[j][r] * rl;
      O[(tok0 + row) * (size_t)CE + coff + j * 16 + ln] = (__bf16)val;
    }
  }
}

// ---------------- host-side orchestration ----------------
extern "C" void kernel_launch(void* const* d_in, const int* in_sizes, int n_in,
                              void* d_out, int out_size, void* d_ws, size_t ws_size,
                              hipStream_t stream) {
  (void)in_sizes; (void)n_in; (void)out_size; (void)ws_size;
  const float* seq   = (const float*)d_in[0];
  const int*   mask  = (const int*)  d_in[1];
  const float* W_emb = (const float*)d_in[2];
  const float* b_emb = (const float*)d_in[3];
  const float* pos   = (const float*)d_in[4];
  const float* Wq = (const float*)d_in[5];  const float* bq = (const float*)d_in[6];
  const float* Wk = (const float*)d_in[7];  const float* bk = (const float*)d_in[8];
  const float* Wv = (const float*)d_in[9];  const float* bv = (const float*)d_in[10];
  const float* Wo = (const float*)d_in[11]; const float* bo = (const float*)d_in[12];
  const float* g1 = (const float*)d_in[13]; const float* be1 = (const float*)d_in[14];
  const float* g2 = (const float*)d_in[15]; const float* be2 = (const float*)d_in[16];
  const float* W1 = (const float*)d_in[17]; const float* b1 = (const float*)d_in[18];
  const float* W2 = (const float*)d_in[19]; const float* b2 = (const float*)d_in[20];
  float* out = (float*)d_out;

  // workspace carve (~84 MB total); all transient, rewritten every call
  char* p = (char*)d_ws;
  auto carve = [&](size_t bytes) -> void* {
    void* r = (void*)p;
    p += (bytes + 255) & ~(size_t)255;
    return r;
  };
  __bf16* seq_bf = (__bf16*)carve((size_t)CT * CF * 2);
  __bf16* wembT  = (__bf16*)carve((size_t)CF * CE * 2);           // [E][F]
  __bf16* wqT    = (__bf16*)carve((size_t)CL * CE * CE * 2);      // [E][E] per layer
  __bf16* wkT    = (__bf16*)carve((size_t)CL * CE * CE * 2);
  __bf16* wvT    = (__bf16*)carve((size_t)CL * CE * CE * 2);
  __bf16* woT    = (__bf16*)carve((size_t)CL * CE * CE * 2);
  __bf16* w1T    = (__bf16*)carve((size_t)CL * CE * CEXP * CE * 2); // [4E][E]
  __bf16* w2T    = (__bf16*)carve((size_t)CL * CEXP * CE * CE * 2); // [E][4E]
  float*  xf     = (float*) carve((size_t)CT * CE * 4);
  __bf16* xb     = (__bf16*)carve((size_t)CT * CE * 2);
  __bf16* qbuf   = (__bf16*)carve((size_t)CT * CE * 2);
  __bf16* kbuf   = (__bf16*)carve((size_t)CT * CE * 2);
  __bf16* vbuf   = (__bf16*)carve((size_t)CT * CE * 2);
  __bf16* vtbuf  = (__bf16*)carve((size_t)CT * CE * 2);
  __bf16* abuf   = (__bf16*)carve((size_t)CT * CE * 2);
  float*  tmpf   = (float*) carve((size_t)CT * CE * 4);
  __bf16* h1b    = (__bf16*)carve((size_t)CT * CEXP * CE * 2);

  // activations: plain convert (row-major)
  {
    const size_t n = (size_t)CT * CF;
    k_cvt_bf16<<<(int)((n / 4 + 255) / 256), 256, 0, stream>>>(seq, seq_bf, (int)n);
  }
  // weights: convert + transpose -> [N][K] bf16
  k_cvt_transpose<<<dim3(CE / 64, CF / 64, 1), 256, 0, stream>>>(W_emb, wembT, CF, CE);
  k_cvt_transpose<<<dim3(CE / 64, CE / 64, CL), 256, 0, stream>>>(Wq, wqT, CE, CE);
  k_cvt_transpose<<<dim3(CE / 64, CE / 64, CL), 256, 0, stream>>>(Wk, wkT, CE, CE);
  k_cvt_transpose<<<dim3(CE / 64, CE / 64, CL), 256, 0, stream>>>(Wv, wvT, CE, CE);
  k_cvt_transpose<<<dim3(CE / 64, CE / 64, CL), 256, 0, stream>>>(Wo, woT, CE, CE);
  k_cvt_transpose<<<dim3(CEXP * CE / 64, CE / 64, CL), 256, 0, stream>>>(W1, w1T, CE, CEXP * CE);
  k_cvt_transpose<<<dim3(CE / 64, CEXP * CE / 64, CL), 256, 0, stream>>>(W2, w2T, CEXP * CE, CE);

  const dim3 gE(CE / GBN, CT / GBM);              // N=512 GEMMs
  const dim3 gF(CEXP * CE / GBN, CT / GBM);       // N=2048 GEMM

  // embedding: tmpf = seq @ W_emb + b_emb ; then + pos_emb
  k_gemm_bf16<false, true, false><<<gE, 256, 0, stream>>>(
      seq_bf, wembT, b_emb, tmpf, (__bf16*)nullptr, CT, CE, CF);
  k_addpos<<<CT, 256, 0, stream>>>(tmpf, pos, xf, xb);

  for (int l = 0; l < CL; ++l) {
    const __bf16* wql = wqT + (size_t)l * CE * CE;
    const __bf16* wkl = wkT + (size_t)l * CE * CE;
    const __bf16* wvl = wvT + (size_t)l * CE * CE;
    const __bf16* wol = woT + (size_t)l * CE * CE;
    const __bf16* w1l = w1T + (size_t)l * CE * CEXP * CE;
    const __bf16* w2l = w2T + (size_t)l * CEXP * CE * CE;

    k_gemm_bf16<false, false, true><<<gE, 256, 0, stream>>>(
        xb, wql, bq + l * CE, (float*)nullptr, qbuf, CT, CE, CE);
    k_gemm_bf16<false, false, true><<<gE, 256, 0, stream>>>(
        xb, wkl, bk + l * CE, (float*)nullptr, kbuf, CT, CE, CE);
    k_gemm_bf16<false, false, true><<<gE, 256, 0, stream>>>(
        xb, wvl, bv + l * CE, (float*)nullptr, vbuf, CT, CE, CE);
    k_vtrans<<<dim3(CS / 64, CB * CH), 256, 0, stream>>>(vbuf, vtbuf);
    k_attn<<<dim3(CS / AQ, CH, CB), 128, 0, stream>>>(qbuf, kbuf, vtbuf, mask, abuf);
    k_gemm_bf16<false, true, false><<<gE, 256, 0, stream>>>(
        abuf, wol, bo + l * CE, tmpf, (__bf16*)nullptr, CT, CE, CE);
    k_ln<<<CT, 256, 0, stream>>>(xf, tmpf, g1 + l * CE, be1 + l * CE, xf, xb);
    k_gemm_bf16<true, false, true><<<gF, 256, 0, stream>>>(
        xb, w1l, b1 + (size_t)l * CEXP * CE, (float*)nullptr, h1b, CT, CEXP * CE, CE);
    k_gemm_bf16<false, true, false><<<gE, 256, 0, stream>>>(
        h1b, w2l, b2 + l * CE, tmpf, (__bf16*)nullptr, CT, CE, CEXP * CE);
    float* lnout = (l == CL - 1) ? out : xf;
    k_ln<<<CT, 256, 0, stream>>>(xf, tmpf, g2 + l * CE, be2 + l * CE, lnout, xb);
  }
}
